// LRSA_89713276879453
// MI455X (gfx1250) — matrix operand order repaired and checked by reading, not compiled, not run on hardware
//
#include <hip/hip_runtime.h>

typedef __attribute__((ext_vector_type(16))) __bf16 v16bf;
typedef __attribute__((ext_vector_type(8)))  float  v8f;
typedef __attribute__((ext_vector_type(8)))  unsigned int v8u;

#define C_DIM 256
#define N_DIM 65536
#define NP    128           // rows of N per block
#define STR   264           // padded bf16 row stride in LDS
#define M_TOK 237
#define SCALE 0.17677669529663687f  // 1/sqrt(32)

// ---------------- helpers ----------------

__device__ __forceinline__ __bf16 f2bf(float f) {
  unsigned u = __builtin_bit_cast(unsigned, f);
  unsigned r = u + 0x7fffu + ((u >> 16) & 1u);
  unsigned short h = (unsigned short)(r >> 16);
  return __builtin_bit_cast(__bf16, h);
}

__device__ __forceinline__ v8f wmma_bf16(v16bf a, v16bf b, v8f c) {
  return __builtin_amdgcn_wmma_f32_16x16x32_bf16(
      /*neg_a=*/false, a, /*neg_b=*/false, b,
      /*c_mod=*/(short)0, c, /*reuse_a=*/false, /*reuse_b=*/false);
}

// A-fragment (16x32 bf16) from LDS row-major tile; p -> &tile[row_of_this_lane][k0]
// layout: lanes0-15 V0..7 K=0..15; lanes16-31 K=16..31
__device__ __forceinline__ v16bf lds_loadA(const __bf16* p, int lane) {
  const unsigned int* q = (const unsigned int*)p;
  int kh = (lane >> 4) * 8;  // in dword units
  v8u t;
#pragma unroll
  for (int j = 0; j < 8; ++j) {
    int ku = kh + j;
    t[j] = q[ku];
  }
  return __builtin_bit_cast(v16bf, t);
}

// B-fragment pre-packed in global: [frag][lane][16 bf16]
__device__ __forceinline__ v16bf g_loadB(const __bf16* frags, int fragIdx, int lane) {
  const v8u* p = (const v8u*)frags;
  v8u t = p[fragIdx * 32 + lane];
  return __builtin_bit_cast(v16bf, t);
}

__device__ __forceinline__ void tok_decode(int m, int& lv, int& p, int& off, int& i, int& j) {
  if (m < 121)      { lv = 0; p = 11; off = 0;   }
  else if (m < 185) { lv = 1; p = 8;  off = 121; }
  else if (m < 221) { lv = 2; p = 6;  off = 185; }
  else              { lv = 3; p = 4;  off = 221; }
  int ml = m - off; i = ml / p; j = ml % p;
}

// ---------------- prep kernels (tiny fp32 path) ----------------

// one block per channel: stage channel in LDS, accumulate adaptive-pool bins
__global__ void pool_kernel(const float* __restrict__ x, float* __restrict__ pooled) {
  extern __shared__ char dynsm[];
  float* rows = (float*)dynsm;           // [256][256]
  float* tok  = rows + 65536;            // [240]
  const int c = blockIdx.x, tid = threadIdx.x;
  const float* xc = x + (size_t)c * N_DIM;
  for (int i = tid; i < 65536; i += 256) rows[i] = xc[i];
  for (int i = tid; i < 240; i += 256) tok[i] = 0.f;
  __syncthreads();
  const int r = tid;
  const int PS[4]  = {11, 8, 6, 4};
  const int OFF[4] = {0, 121, 185, 221};
#pragma unroll
  for (int lv = 0; lv < 4; ++lv) {
    int p = PS[lv];
    for (int b = 0; b < p; ++b) {
      int cs = (b * 256) / p, ce = ((b + 1) * 256 + p - 1) / p;
      float s = 0.f;
      for (int w = cs; w < ce; ++w) s += rows[r * 256 + w];
      int rb = (r * p) >> 8;                       // main row-bin (bins may overlap)
      atomicAdd(&tok[OFF[lv] + rb * p + b], s);
      if (rb > 0 && r < (rb * 256 + p - 1) / p)
        atomicAdd(&tok[OFF[lv] + (rb - 1) * p + b], s);
      if (rb < p - 1 && r >= ((rb + 1) * 256) / p)
        atomicAdd(&tok[OFF[lv] + (rb + 1) * p + b], s);
    }
  }
  __syncthreads();
  for (int m = tid; m < M_TOK; m += 256) {
    int lv, p, off, i, j; tok_decode(m, lv, p, off, i, j);
    int rs = (i * 256) / p, re = ((i + 1) * 256 + p - 1) / p;
    int qs = (j * 256) / p, qe = ((j + 1) * 256 + p - 1) / p;
    pooled[c * M_TOK + m] = tok[m] / (float)((re - rs) * (qe - qs));
  }
}

// depthwise 3x3 + residual; one block per token, thread = channel
__global__ void dconv_kernel(const float* __restrict__ pooled, const float* __restrict__ dw,
                             const float* __restrict__ db, float* __restrict__ t) {
  const int m = blockIdx.x, c = threadIdx.x;
  int lv, p, off, i, j; tok_decode(m, lv, p, off, i, j);
  float center = pooled[c * M_TOK + m];
  float acc = db[lv * C_DIM + c];
#pragma unroll
  for (int di = -1; di <= 1; ++di)
#pragma unroll
    for (int dj = -1; dj <= 1; ++dj) {
      int ii = i + di, jj = j + dj;
      if (ii >= 0 && ii < p && jj >= 0 && jj < p) {
        float w = dw[(size_t)(lv * C_DIM + c) * 9 + (di + 1) * 3 + (dj + 1)];
        acc += w * pooled[c * M_TOK + off + ii * p + jj];
      }
    }
  t[m * C_DIM + c] = center + acc;
}

__global__ void ln_kernel(const float* __restrict__ t, const float* __restrict__ g,
                          const float* __restrict__ b, float* __restrict__ lnout) {
  __shared__ float red[256];
  const int m = blockIdx.x, tid = threadIdx.x;
  float v = t[m * C_DIM + tid];
  red[tid] = v; __syncthreads();
  for (int s2 = 128; s2 > 0; s2 >>= 1) { if (tid < s2) red[tid] += red[tid + s2]; __syncthreads(); }
  float mu = red[0] / 256.f; __syncthreads();
  float d = v - mu;
  red[tid] = d * d; __syncthreads();
  for (int s2 = 128; s2 > 0; s2 >>= 1) { if (tid < s2) red[tid] += red[tid + s2]; __syncthreads(); }
  float var = red[0] / 256.f;
  lnout[m * C_DIM + tid] = d * rsqrtf(var + 1e-5f) * g[tid] + b[tid];
}

__global__ void kv_kernel(const float* __restrict__ lnin, const float* __restrict__ Wkv,
                          float* __restrict__ kv) {
  __shared__ float row[256];
  const int m = blockIdx.x, tid = threadIdx.x;
  row[tid] = lnin[m * C_DIM + tid];
  __syncthreads();
  float a0 = 0.f, a1 = 0.f;
  for (int c2 = 0; c2 < C_DIM; ++c2) {
    float v = row[c2];
    a0 += v * Wkv[(size_t)tid * C_DIM + c2];
    a1 += v * Wkv[(size_t)(C_DIM + tid) * C_DIM + c2];
  }
  kv[m * 512 + tid] = a0;
  kv[m * 512 + 256 + tid] = a1;
}

// ---------------- fragment packers (B-matrix layout, bf16) ----------------

// W[256][256] row-major; B[k=c][col=o] = W[o][c]; frag(ct,kc)
__global__ void pack_w_kernel(const float* __restrict__ W, __bf16* __restrict__ dst) {
  int idx = blockIdx.x * 256 + threadIdx.x;          // 65536
  int e = idx & 15, lane = (idx >> 4) & 31, kc = (idx >> 9) & 7, ct = idx >> 12;
  int col  = ct * 16 + (lane & 15);
  int krow = kc * 32 + ((lane >> 4) << 4) + e;
  dst[idx] = f2bf(W[(size_t)col * C_DIM + krow]);
}

// K^T fragments: frag(hd,mt): B[dd][m] = kv[m][hd*32+dd], m>=237 -> 0
__global__ void pack_k_kernel(const float* __restrict__ kv, __bf16* __restrict__ dst) {
  int idx = blockIdx.x * 256 + threadIdx.x;          // 65536
  int e = idx & 15, lane = (idx >> 4) & 31, mt = (idx >> 9) & 15, hd = idx >> 13;
  int m  = mt * 16 + (lane & 15);
  int dd = ((lane >> 4) << 4) + e;
  dst[idx] = (m < M_TOK) ? f2bf(kv[m * 512 + hd * 32 + dd]) : f2bf(0.f);
}

// V fragments: frag(hd,ct,kc): B[m][dd] = kv[m][256 + hd*32 + ct*16 + col]
__global__ void pack_v_kernel(const float* __restrict__ kv, __bf16* __restrict__ dst) {
  int idx = blockIdx.x * 256 + threadIdx.x;          // 65536
  int e = idx & 15, lane = (idx >> 4) & 31, kc = (idx >> 9) & 7;
  int ct = (idx >> 12) & 1, hd = idx >> 13;
  int m  = kc * 32 + ((lane >> 4) << 4) + e;
  int dd = ct * 16 + (lane & 15);
  dst[idx] = (m < M_TOK) ? f2bf(kv[m * 512 + 256 + hd * 32 + dd]) : f2bf(0.f);
}

// ---------------- fused main kernel: Q-gemm + attention + proj ----------------

__launch_bounds__(256, 1)
__global__ void attn_main_kernel(const float* __restrict__ x,
                                 const __bf16* __restrict__ WqF,
                                 const __bf16* __restrict__ WpF,
                                 const __bf16* __restrict__ Kf,
                                 const __bf16* __restrict__ Vf,
                                 const float* __restrict__ bproj,
                                 float* __restrict__ out) {
  extern __shared__ char dynsm[];
  __bf16* xs = (__bf16*)dynsm;        // [NP][STR] x-tile (bf16), reused as attn-out
  __bf16* qs = xs + NP * STR;         // [NP][STR] q (scaled, bf16)
  __bf16* ps = qs + NP * STR;         // [8 waves][16][STR] probs transpose buffer

  const int tid  = threadIdx.x;
  const int lane = tid & 31;
  const int wave = tid >> 5;
  const int n0   = blockIdx.x * NP;

  // stage x-tile transposed to [n][c] bf16 (coalesced over n)
  for (int idx = tid; idx < C_DIM * NP; idx += 256) {
    int c = idx >> 7;
    int nn = idx & (NP - 1);
    xs[nn * STR + c] = f2bf(x[(size_t)c * N_DIM + n0 + nn]);
  }
  __syncthreads();   // only block-wide sync needed; everything below is wave-private rows

  const int r0    = wave * 16;
  const int rowA  = r0 + (lane & 15);
  const int halfo = (lane >> 4) * 8;

  // ---- Q = x @ Wq^T, scaled by 1/sqrt(d) ----
  v16bf aq[8];
#pragma unroll
  for (int kc = 0; kc < 8; ++kc)
    aq[kc] = lds_loadA(&xs[rowA * STR + kc * 32], lane);

#pragma unroll 1
  for (int ct = 0; ct < 16; ++ct) {
    v8f acc = {0.f, 0.f, 0.f, 0.f, 0.f, 0.f, 0.f, 0.f};
#pragma unroll
    for (int kc = 0; kc < 8; ++kc)
      acc = wmma_bf16(aq[kc], g_loadB(WqF, ct * 8 + kc, lane), acc);
    int col = ct * 16 + (lane & 15);
#pragma unroll
    for (int r = 0; r < 8; ++r)
      qs[(r0 + halfo + r) * STR + col] = f2bf(acc[r] * SCALE);
  }

  __bf16* pw = ps + wave * 16 * STR;

  // ---- per-head flash-style attention over M=237 (padded 256) ----
#pragma unroll 1
  for (int hd = 0; hd < 8; ++hd) {
    v16bf aqh = lds_loadA(&qs[rowA * STR + hd * 32], lane);
    v8f s[16];
#pragma unroll
    for (int mt = 0; mt < 16; ++mt) {
      v8f z = {0.f, 0.f, 0.f, 0.f, 0.f, 0.f, 0.f, 0.f};
      s[mt] = wmma_bf16(aqh, g_loadB(Kf, hd * 16 + mt, lane), z);
    }
    // softmax per output row (rows = halfo + r for this lane's half)
    float rinv[8];
#pragma unroll
    for (int r = 0; r < 8; ++r) {
      float mx = -1e30f;
#pragma unroll
      for (int mt = 0; mt < 16; ++mt) {
        int m = mt * 16 + (lane & 15);
        float v = (m < M_TOK) ? s[mt][r] : -1e30f;
        mx = fmaxf(mx, v);
      }
#pragma unroll
      for (int d2 = 1; d2 < 16; d2 <<= 1)
        mx = fmaxf(mx, __shfl_xor(mx, d2, 32));
      float sum = 0.f;
#pragma unroll
      for (int mt = 0; mt < 16; ++mt) {
        int m = mt * 16 + (lane & 15);
        float pr = (m < M_TOK) ? __expf(s[mt][r] - mx) : 0.f;
        s[mt][r] = pr;
        sum += pr;
      }
#pragma unroll
      for (int d2 = 1; d2 < 16; d2 <<= 1)
        sum += __shfl_xor(sum, d2, 32);
      rinv[r] = 1.f / sum;
    }
    // transpose probs D-layout -> A-layout via per-wave LDS
#pragma unroll
    for (int mt = 0; mt < 16; ++mt) {
      int mcol = mt * 16 + (lane & 15);
#pragma unroll
      for (int r = 0; r < 8; ++r)
        pw[(halfo + r) * STR + mcol] = f2bf(s[mt][r]);
    }
    // PV: out[n][hd*32 + ct2*16 + col] (normalize by rowsum here)
#pragma unroll
    for (int ct2 = 0; ct2 < 2; ++ct2) {
      v8f o = {0.f, 0.f, 0.f, 0.f, 0.f, 0.f, 0.f, 0.f};
#pragma unroll
      for (int kc = 0; kc < 8; ++kc) {
        v16bf ap = lds_loadA(&pw[(lane & 15) * STR + kc * 32], lane);
        o = wmma_bf16(ap, g_loadB(Vf, (hd * 2 + ct2) * 8 + kc, lane), o);
      }
      int col = hd * 32 + ct2 * 16 + (lane & 15);
#pragma unroll
      for (int r = 0; r < 8; ++r)
        xs[(r0 + halfo + r) * STR + col] = f2bf(o[r] * rinv[r]);
    }
  }

  // ---- out = attn_out @ Wproj^T + bproj, stored [co][n] ----
  v16bf ao[8];
#pragma unroll
  for (int kc = 0; kc < 8; ++kc)
    ao[kc] = lds_loadA(&xs[rowA * STR + kc * 32], lane);
#pragma unroll 1
  for (int ct = 0; ct < 16; ++ct) {
    v8f acc = {0.f, 0.f, 0.f, 0.f, 0.f, 0.f, 0.f, 0.f};
#pragma unroll
    for (int kc = 0; kc < 8; ++kc)
      acc = wmma_bf16(ao[kc], g_loadB(WpF, ct * 8 + kc, lane), acc);
    int co = ct * 16 + (lane & 15);
    float bb = bproj[co];
    int nb = n0 + r0 + halfo;
#pragma unroll
    for (int r = 0; r < 8; ++r)
      out[(size_t)co * N_DIM + nb + r] = acc[r] + bb;
  }
}

// ---------------- launch ----------------

extern "C" void kernel_launch(void* const* d_in, const int* in_sizes, int n_in,
                              void* d_out, int out_size, void* d_ws, size_t ws_size,
                              hipStream_t stream) {
  (void)in_sizes; (void)n_in; (void)out_size; (void)ws_size;
  const float* x     = (const float*)d_in[0];
  const float* Wq    = (const float*)d_in[1];
  const float* Wkv   = (const float*)d_in[2];
  const float* Wproj = (const float*)d_in[3];
  const float* bproj = (const float*)d_in[4];
  const float* lng   = (const float*)d_in[5];
  const float* lnb   = (const float*)d_in[6];
  const float* dw    = (const float*)d_in[7];
  const float* db    = (const float*)d_in[8];
  float* out = (float*)d_out;

  char* ws = (char*)d_ws;
  size_t off = 0;
  auto take = [&](size_t bytes) -> void* {
    void* p = ws + off;
    off += (bytes + 511) & ~(size_t)511;
    return p;
  };
  float*  pooled = (float*)take(256 * M_TOK * sizeof(float));
  float*  tbuf   = (float*)take(M_TOK * 256 * sizeof(float));
  float*  lnbuf  = (float*)take(M_TOK * 256 * sizeof(float));
  float*  kvbuf  = (float*)take(M_TOK * 512 * sizeof(float));
  __bf16* WqF    = (__bf16*)take(65536 * 2);
  __bf16* WpF    = (__bf16*)take(65536 * 2);
  __bf16* Kf     = (__bf16*)take(65536 * 2);
  __bf16* Vf     = (__bf16*)take(65536 * 2);

  const int POOL_LDS = 65536 * 4 + 240 * 4;     // 263104 B
  const int MAIN_LDS = 3 * NP * STR * 2;        // 202752 B
  hipFuncSetAttribute((const void*)pool_kernel,
                      hipFuncAttributeMaxDynamicSharedMemorySize, POOL_LDS);
  hipFuncSetAttribute((const void*)attn_main_kernel,
                      hipFuncAttributeMaxDynamicSharedMemorySize, MAIN_LDS);

  pool_kernel<<<256, 256, POOL_LDS, stream>>>(x, pooled);
  dconv_kernel<<<M_TOK, 256, 0, stream>>>(pooled, dw, db, tbuf);
  ln_kernel<<<M_TOK, 256, 0, stream>>>(tbuf, lng, lnb, lnbuf);
  kv_kernel<<<M_TOK, 256, 0, stream>>>(lnbuf, Wkv, kvbuf);
  pack_w_kernel<<<256, 256, 0, stream>>>(Wq, WqF);
  pack_w_kernel<<<256, 256, 0, stream>>>(Wproj, WpF);
  pack_k_kernel<<<256, 256, 0, stream>>>(kvbuf, Kf);
  pack_v_kernel<<<256, 256, 0, stream>>>(kvbuf, Vf);
  attn_main_kernel<<<512, 256, MAIN_LDS, stream>>>(x, WqF, WpF, Kf, Vf, bproj, out);
}
